// Step_343597384084
// MI455X (gfx1250) — compile-verified, matrix operand order checked by reading, not run on hardware
//
#include <hip/hip_runtime.h>

// ---------------------------------------------------------------------------
// Problem constants (reference: B=4, L=4096, D=1024, fp32)
// ---------------------------------------------------------------------------
constexpr int Dd     = 1024;
constexpr int Ll     = 4096;
constexpr int Bb     = 4;
constexpr int Mm     = Bb * Ll;          // 16384 rows
constexpr int NCHUNK = 64;               // scan chunks per sequence
constexpr int CLEN   = Ll / NCHUNK;      // 64 steps per chunk
constexpr int KTILES = Dd / 32;

typedef __attribute__((ext_vector_type(16))) __bf16 v16bf;
typedef __attribute__((ext_vector_type(8)))  float  v8f;

union AFrag {
    v16bf v;
    unsigned short u[16];
    uint4 q[2];
};

__device__ __forceinline__ unsigned short f2bf(float f) {
    union { float f; unsigned int u; } x; x.f = f;
    unsigned int r = x.u + 0x7FFFu + ((x.u >> 16) & 1u);   // round-to-nearest-even
    return (unsigned short)(r >> 16);
}

__device__ __forceinline__ float sigm(float x)  { return 1.0f / (1.0f + __expf(-x)); }
__device__ __forceinline__ float silu(float x)  { return x / (1.0f + __expf(-x)); }

// ---------------------------------------------------------------------------
// gfx1250 async global->LDS copy (ASYNCcnt-tracked), with sync fallback
// Builtin signature (from hipcc diagnostic): param1 = int4 AS(1)*, param2 = AS(3)*
// ---------------------------------------------------------------------------
#if defined(__gfx1250__) && __has_builtin(__builtin_amdgcn_global_load_async_to_lds_b128)
#define ASYNC_COPY_OK 1
#else
#define ASYNC_COPY_OK 0
#endif

typedef int ivec4 __attribute__((vector_size(16)));

#define AS1P(p) ((__attribute__((address_space(1))) ivec4*)(unsigned long long)(p))
#define AS3P(p) ((__attribute__((address_space(3))) ivec4*)(unsigned int)(unsigned long long)(p))

#if ASYNC_COPY_OK
#  if __has_builtin(__builtin_amdgcn_s_wait_asynccnt)
#    define ASYNC_WAIT(n) __builtin_amdgcn_s_wait_asynccnt(n)
#  else
#    define ASYNC_WAIT(n) asm volatile("s_wait_asynccnt %0" :: "i"(n) : "memory")
#  endif
#else
#  define ASYNC_WAIT(n) ((void)0)
#endif

#define LDS_STRIDE 40   // 80B rows: 16B-aligned, odd 16B-unit stride (20 dwords) -> conflict-free
typedef unsigned short TileT[128][LDS_STRIDE];

// Stage a 128(row) x 32(col) bf16 tile from row-major src (row stride Dd) into LDS.
// 512 16-byte chunks, 256 threads -> 2 chunks (2 async instructions) per thread.
__device__ __forceinline__ void stage128x32(const unsigned short* __restrict__ src,
                                            int row0, int col0, TileT dst, int tid)
{
    #pragma unroll
    for (int j = 0; j < 2; ++j) {
        int i  = tid + j * 256;
        int r  = i >> 2;
        int c8 = (i & 3) * 8;
        const unsigned short* g = src + (size_t)(row0 + r) * Dd + (col0 + c8);
        unsigned short* l = &dst[r][c8];
#if ASYNC_COPY_OK
        __builtin_amdgcn_global_load_async_to_lds_b128(AS1P(g), AS3P(l), 0, 0);
#else
        *(uint4*)l = *(const uint4*)g;
#endif
    }
}

__device__ __forceinline__ void load_frag(AFrag& f, const unsigned short (*tile)[LDS_STRIDE],
                                          int row, int khalf)
{
    f.q[0] = *(const uint4*)&tile[row][khalf];
    f.q[1] = *(const uint4*)&tile[row][16 + khalf];
}

// ---------------------------------------------------------------------------
// Kernel 0: tiled transpose + fp32->bf16 convert of one weight matrix:
//           Wt[n][k] = bf16(W[k][n])   (so GEMM B-tiles are contiguous in K)
// ---------------------------------------------------------------------------
__global__ __launch_bounds__(256) void transpose_convert(
    const float* __restrict__ W, unsigned short* __restrict__ Wt)
{
    __shared__ float tile[32][33];
    const int bx = blockIdx.x * 32;     // n range
    const int by = blockIdx.y * 32;     // k range
    const int tx = threadIdx.x, ty = threadIdx.y;   // (32, 8)
    #pragma unroll
    for (int j = 0; j < 4; ++j)
        tile[ty + j * 8][tx] = W[(size_t)(by + ty + j * 8) * Dd + bx + tx];
    __syncthreads();
    #pragma unroll
    for (int j = 0; j < 4; ++j)
        Wt[(size_t)(bx + ty + j * 8) * Dd + by + tx] = f2bf(tile[tx][ty + j * 8]);
}

// ---------------------------------------------------------------------------
// Kernel 1: RMS-norm of ctx -> bf16 activations (one block per row)
// ---------------------------------------------------------------------------
__global__ __launch_bounds__(256) void rmsnorm_bf16(
    const float* __restrict__ ctx, const float* __restrict__ g_rms,
    unsigned short* __restrict__ ctxn)
{
    __shared__ float red[256];
    const int tid = threadIdx.x;
    const size_t base = (size_t)blockIdx.x * Dd;

    float x[4];
    float acc = 0.f;
    #pragma unroll
    for (int j = 0; j < 4; ++j) {
        x[j] = ctx[base + tid + j * 256];
        acc += x[j] * x[j];
    }
    red[tid] = acc;
    __syncthreads();
    for (int s = 128; s > 0; s >>= 1) {
        if (tid < s) red[tid] += red[tid + s];
        __syncthreads();
    }
    const float rinv = rsqrtf(red[0] * (1.0f / Dd) + 1e-6f);
    #pragma unroll
    for (int j = 0; j < 4; ++j) {
        int d = tid + j * 256;
        ctxn[base + d] = f2bf(x[j] * rinv * g_rms[d]);
    }
}

// ---------------------------------------------------------------------------
// Kernel 2: fused dual GEMM (alpha & beta share A = ctx_norm), async double-buffer
//   block tile 128x128, 8 waves, wave tile 64x32 (4x2 frags), K step 32
// ---------------------------------------------------------------------------
__global__ __launch_bounds__(256) void gemm_gates(
    const unsigned short* __restrict__ Abf,      // [M,D] bf16 ctx_norm
    const unsigned short* __restrict__ WaT,      // [D,D] bf16, transposed: [n][k]
    const unsigned short* __restrict__ WbT,      // [D,D] bf16, transposed
    const float* __restrict__ b_alpha, const float* __restrict__ b_beta,
    const float* __restrict__ v,
    float* __restrict__ alphas, float* __restrict__ u_out)
{
    __shared__ alignas(16) unsigned short As [2][128][LDS_STRIDE];
    __shared__ alignas(16) unsigned short Bta[2][128][LDS_STRIDE];
    __shared__ alignas(16) unsigned short Btb[2][128][LDS_STRIDE];

    const int tid  = threadIdx.x;
    const int m0   = blockIdx.y * 128;
    const int n0   = blockIdx.x * 128;
    const int wave = tid >> 5, lane = tid & 31;
    const int wm = wave >> 2, wn = wave & 3;       // 2 x 4 wave grid
    const int lm = lane & 15, lhi = lane >> 4;
    const int khalf = lhi * 8;

    v8f cA[4][2] = {};
    v8f cB[4][2] = {};

    // prologue: stage K-tile 0 into buffer 0 (6 async ops per wave)
    stage128x32(Abf, m0, 0, As [0], tid);
    stage128x32(WaT, n0, 0, Bta[0], tid);
    stage128x32(WbT, n0, 0, Btb[0], tid);

    for (int kt = 0; kt < KTILES; ++kt) {
        const int cur = kt & 1;
        if (kt + 1 < KTILES) {
            const int nxt = (kt + 1) & 1;
            const int k0n = (kt + 1) * 32;
            stage128x32(Abf, m0, k0n, As [nxt], tid);
            stage128x32(WaT, n0, k0n, Bta[nxt], tid);
            stage128x32(WbT, n0, k0n, Btb[nxt], tid);
            ASYNC_WAIT(6);            // in-order completion => current buffer ready
        } else {
            ASYNC_WAIT(0);
        }
        __syncthreads();

        AFrag a[4], ba[2], bb[2];
        #pragma unroll
        for (int fm = 0; fm < 4; ++fm) load_frag(a[fm], As[cur], wm * 64 + fm * 16 + lm, khalf);
        #pragma unroll
        for (int fn = 0; fn < 2; ++fn) {
            load_frag(ba[fn], Bta[cur], wn * 32 + fn * 16 + lm, khalf);
            load_frag(bb[fn], Btb[cur], wn * 32 + fn * 16 + lm, khalf);
        }
        #pragma unroll
        for (int fm = 0; fm < 4; ++fm)
            #pragma unroll
            for (int fn = 0; fn < 2; ++fn) {
                cA[fm][fn] = __builtin_amdgcn_wmma_f32_16x16x32_bf16(
                    false, a[fm].v, false, ba[fn].v, (short)0, cA[fm][fn], false, false);
                cB[fm][fn] = __builtin_amdgcn_wmma_f32_16x16x32_bf16(
                    false, a[fm].v, false, bb[fn].v, (short)0, cB[fm][fn], false, false);
            }
        __syncthreads();
    }

    #pragma unroll
    for (int fm = 0; fm < 4; ++fm)
        #pragma unroll
        for (int fn = 0; fn < 2; ++fn)
            #pragma unroll
            for (int r = 0; r < 8; ++r) {
                int m = m0 + wm * 64 + fm * 16 + r + (lhi ? 8 : 0);
                int n = n0 + wn * 32 + fn * 16 + lm;
                size_t idx = (size_t)m * Dd + n;
                float alpha = sigm(cA[fm][fn][r] + b_alpha[n]);
                float beta  = silu(cB[fm][fn][r] + b_beta[n]);
                float wsc   = sqrtf(fmaxf(1.0f - alpha * alpha, 1e-6f));
                alphas[idx] = alpha;
                u_out[idx]  = v[idx] * beta * wsc;
            }
}

// ---------------------------------------------------------------------------
// Kernel 5: GEMM3  ctx_out = ctx + silu(fetched @ W_ctx + b_ctx)
//   A = fetched in bf16 (written by scan phase 3), async double-buffer
// ---------------------------------------------------------------------------
__global__ __launch_bounds__(256) void gemm_ctx(
    const unsigned short* __restrict__ fetbf,    // [M,D] bf16
    const unsigned short* __restrict__ WcT,      // [D,D] bf16, transposed
    const float* __restrict__ b_ctx,
    const float* __restrict__ ctx_in,
    float* __restrict__ ctx_out)
{
    __shared__ alignas(16) unsigned short As [2][128][LDS_STRIDE];
    __shared__ alignas(16) unsigned short Btc[2][128][LDS_STRIDE];

    const int tid  = threadIdx.x;
    const int m0   = blockIdx.y * 128;
    const int n0   = blockIdx.x * 128;
    const int wave = tid >> 5, lane = tid & 31;
    const int wm = wave >> 2, wn = wave & 3;
    const int lm = lane & 15, lhi = lane >> 4;
    const int khalf = lhi * 8;

    v8f c[4][2] = {};

    stage128x32(fetbf, m0, 0, As [0], tid);
    stage128x32(WcT,   n0, 0, Btc[0], tid);

    for (int kt = 0; kt < KTILES; ++kt) {
        const int cur = kt & 1;
        if (kt + 1 < KTILES) {
            const int nxt = (kt + 1) & 1;
            const int k0n = (kt + 1) * 32;
            stage128x32(fetbf, m0, k0n, As [nxt], tid);
            stage128x32(WcT,   n0, k0n, Btc[nxt], tid);
            ASYNC_WAIT(4);
        } else {
            ASYNC_WAIT(0);
        }
        __syncthreads();

        AFrag a[4], bc[2];
        #pragma unroll
        for (int fm = 0; fm < 4; ++fm) load_frag(a[fm], As[cur], wm * 64 + fm * 16 + lm, khalf);
        #pragma unroll
        for (int fn = 0; fn < 2; ++fn) load_frag(bc[fn], Btc[cur], wn * 32 + fn * 16 + lm, khalf);

        #pragma unroll
        for (int fm = 0; fm < 4; ++fm)
            #pragma unroll
            for (int fn = 0; fn < 2; ++fn)
                c[fm][fn] = __builtin_amdgcn_wmma_f32_16x16x32_bf16(
                    false, a[fm].v, false, bc[fn].v, (short)0, c[fm][fn], false, false);
        __syncthreads();
    }

    #pragma unroll
    for (int fm = 0; fm < 4; ++fm)
        #pragma unroll
        for (int fn = 0; fn < 2; ++fn)
            #pragma unroll
            for (int r = 0; r < 8; ++r) {
                int m = m0 + wm * 64 + fm * 16 + r + (lhi ? 8 : 0);
                int n = n0 + wn * 32 + fn * 16 + lm;
                size_t idx = (size_t)m * Dd + n;
                ctx_out[idx] = ctx_in[idx] + silu(c[fm][fn][r] + b_ctx[n]);
            }
}

// ---------------------------------------------------------------------------
// Chunked linear-recurrence scan:  h_t = a_t * h_{t-1} + x_t  over L per (b,d)
// ---------------------------------------------------------------------------
__global__ __launch_bounds__(256) void scan_phase1(
    const float* __restrict__ u, const float* __restrict__ a,
    float* __restrict__ chunkH, float* __restrict__ chunkA)
{
    int g = blockIdx.x * blockDim.x + threadIdx.x;     // over B*NCHUNK*D
    int d = g & (Dd - 1);
    int c = (g >> 10) & (NCHUNK - 1);
    int b = g >> 16;
    size_t base = ((size_t)b * Ll + (size_t)c * CLEN) * Dd + d;
    float h = 0.f, A = 1.f;
    for (int t = 0; t < CLEN; ++t) {
        float at = a[base + (size_t)t * Dd];
        float xt = u[base + (size_t)t * Dd];
        h = at * h + xt;
        A *= at;
    }
    chunkH[g] = h;
    chunkA[g] = A;
}

__global__ __launch_bounds__(256) void scan_phase2(
    const float* __restrict__ chunkH, const float* __restrict__ chunkA,
    float* __restrict__ prefix)
{
    int g = blockIdx.x * blockDim.x + threadIdx.x;     // over B*D
    int d = g & (Dd - 1);
    int b = g >> 10;
    float run = 0.f;
    for (int c = 0; c < NCHUNK; ++c) {
        int idx = (b * NCHUNK + c) * Dd + d;
        prefix[idx] = run;                              // exclusive prefix before chunk c
        run = chunkA[idx] * run + chunkH[idx];
    }
}

__global__ __launch_bounds__(256) void scan_phase3(
    const float* __restrict__ u, const float* __restrict__ a,
    const float* __restrict__ prefix,
    const float* __restrict__ out_in,
    unsigned short* __restrict__ fetched_bf,            // bf16 for GEMM3 A operand
    float* __restrict__ out_sec)
{
    int g = blockIdx.x * blockDim.x + threadIdx.x;     // over B*NCHUNK*D
    int d = g & (Dd - 1);
    int c = (g >> 10) & (NCHUNK - 1);
    int b = g >> 16;
    size_t base = ((size_t)b * Ll + (size_t)c * CLEN) * Dd + d;
    float h = prefix[g];
    for (int t = 0; t < CLEN; ++t) {
        size_t idx = base + (size_t)t * Dd;
        h = a[idx] * h + u[idx];
        fetched_bf[idx] = f2bf(h);
        out_sec[idx] = out_in[idx] + h;
    }
}

// ---------------------------------------------------------------------------
// Host-side orchestration
// ---------------------------------------------------------------------------
extern "C" void kernel_launch(void* const* d_in, const int* in_sizes, int n_in,
                              void* d_out, int out_size, void* d_ws, size_t ws_size,
                              hipStream_t stream) {
    (void)in_sizes; (void)n_in; (void)out_size; (void)ws_size;

    const float* v            = (const float*)d_in[0];
    const float* ctx          = (const float*)d_in[1];
    const float* out_in       = (const float*)d_in[2];
    const float* alpha_logits = (const float*)d_in[3];
    const float* g_rms        = (const float*)d_in[4];
    const float* W_alpha      = (const float*)d_in[5];
    const float* b_alpha      = (const float*)d_in[6];
    const float* W_beta       = (const float*)d_in[7];
    const float* b_beta       = (const float*)d_in[8];
    const float* W_ctx        = (const float*)d_in[9];
    const float* b_ctx        = (const float*)d_in[10];

    const size_t SEC = (size_t)Mm * Dd;                // 16,777,216 elements
    float* out_v   = (float*)d_out;
    float* out_ctx = out_v + SEC;
    float* out_out = out_ctx + SEC;
    float* out_al  = out_out + SEC;

    // workspace carve-up
    char* w = (char*)d_ws;
    unsigned short* ctxn = (unsigned short*)(w);                          // 32 MB
    unsigned short* WaT  = (unsigned short*)(w + SEC * 2);                //  2 MB
    unsigned short* WbT  = (unsigned short*)(w + SEC * 2 + (size_t)Dd * Dd * 2);
    unsigned short* WcT  = (unsigned short*)(w + SEC * 2 + (size_t)Dd * Dd * 4);
    float* alphas  = (float*)(w + SEC * 2 + (size_t)Dd * Dd * 6);         // 64 MB
    float* u_buf   = alphas + SEC;                                        // 64 MB
    unsigned short* fetbf = (unsigned short*)(u_buf + SEC);               // 32 MB
    float* chunkH  = (float*)(fetbf + SEC);                               //  1 MB each
    float* chunkA  = chunkH + (size_t)Bb * NCHUNK * Dd;
    float* prefix  = chunkA + (size_t)Bb * NCHUNK * Dd;

    // 0) weights -> bf16, transposed so B tiles are K-contiguous
    dim3 tgrid(Dd / 32, Dd / 32), tblk(32, 8);
    transpose_convert<<<tgrid, tblk, 0, stream>>>(W_alpha, WaT);
    transpose_convert<<<tgrid, tblk, 0, stream>>>(W_beta,  WbT);
    transpose_convert<<<tgrid, tblk, 0, stream>>>(W_ctx,   WcT);

    // 1) rms-norm -> bf16 activations
    rmsnorm_bf16<<<Mm, 256, 0, stream>>>(ctx, g_rms, ctxn);

    // 2) fused alpha/beta GEMMs + gate epilogue
    gemm_gates<<<dim3(Dd / 128, Mm / 128), 256, 0, stream>>>(
        ctxn, WaT, WbT, b_alpha, b_beta, v, alphas, u_buf);

    // 3) chunked scan
    const int scan_threads = Bb * NCHUNK * Dd;         // 262144
    scan_phase1<<<scan_threads / 256, 256, 0, stream>>>(u_buf, alphas, chunkH, chunkA);
    scan_phase2<<<(Bb * Dd) / 256, 256, 0, stream>>>(chunkH, chunkA, prefix);
    scan_phase3<<<scan_threads / 256, 256, 0, stream>>>(
        u_buf, alphas, prefix, out_in, fetbf, out_out);

    // 4) ctx GEMM + silu residual
    gemm_ctx<<<dim3(Dd / 128, Mm / 128), 256, 0, stream>>>(
        fetbf, WcT, b_ctx, ctx, out_ctx);

    // 5) passthrough outputs
    (void)hipMemcpyAsync(out_v,  v,            SEC * sizeof(float), hipMemcpyDeviceToDevice, stream);
    (void)hipMemcpyAsync(out_al, alpha_logits, SEC * sizeof(float), hipMemcpyDeviceToDevice, stream);
}
